// Model_58042188038409
// MI455X (gfx1250) — compile-verified, matrix operand order checked by reading, not run on hardware
//
#include <hip/hip_runtime.h>
#include <hip/hip_bf16.h>
#include <math.h>

typedef __attribute__((ext_vector_type(16))) _Float16 v16h;
typedef __attribute__((ext_vector_type(8)))  _Float16 v8h;
typedef __attribute__((ext_vector_type(8)))  float    v8f;

// ---- model dims ----
static const int BB    = 256;    // batch
static const int T     = 14;     // qlen max
static const int BT    = 3584;   // B*T
static const int EMB   = 300, EMBP = 320;
static const int HID   = 1024, H3 = 3072;
static const int FEAT  = 2052, FEATP = 2080;
static const int KOBJ  = 36;
static const int NS    = 10;
static const int BNS   = 2560;   // B*NS
static const int NK    = 8;
static const int GN    = 3076, GNP = 3104;
static const int OUT   = 3000, OUTP = 3008;

#define GSTRIDE(i, n) for (size_t i = blockIdx.x*(size_t)blockDim.x + threadIdx.x; \
                           i < (size_t)(n); i += (size_t)gridDim.x*blockDim.x)

// Async-to-LDS path only if the gfx1250 builtins exist on this toolchain.
#if defined(__AMDGCN__) && __has_builtin(__builtin_amdgcn_global_load_async_to_lds_b128) && \
    __has_builtin(__builtin_amdgcn_s_wait_asynccnt)
#define USE_ASYNC_LDS 1
// builtin signature: (v4i32 AS1*, v4i32 AS3*, imm offset, imm cpol)
typedef int v4i32 __attribute__((vector_size(16)));
typedef __attribute__((address_space(1))) v4i32 g_v4i32;
typedef __attribute__((address_space(3))) v4i32 l_v4i32;
#else
#define USE_ASYNC_LDS 0
#endif

// =====================================================================
// WMMA GEMM:  C[M,N] = act( A[M,K](f16) * W[N,K](f16)^T + bias[N] )
// Block tile 128x64 (4 waves, each 64x32 = 4x2 WMMA tiles).
// M % 128 == 0, N % 64 == 0, K % 32 == 0; rows 16B aligned.
// =====================================================================
__device__ __forceinline__ v16h load_frag(const _Float16* __restrict__ base,
                                          int ld, int lane) {
  // 16-bit fragment layout (CDNA5 ISA 7.12.2):
  // lanes 0-15 : row r=lane,    halves = K[0..7]  and K[16..23]
  // lanes 16-31: row r=lane-16, halves = K[8..15] and K[24..31]
  int r  = lane & 15;
  int hs = lane >> 4;
  const _Float16* p = base + (size_t)r * ld + hs * 8;
  v8h lo = *(const v8h*)(p);
  v8h hi = *(const v8h*)(p + 16);
  v16h o;
#pragma unroll
  for (int i = 0; i < 8; ++i) { o[i] = lo[i]; o[i + 8] = hi[i]; }
  return o;
}

#define LDSA_LD 40   // padded LDS row pitch (halves) to spread banks
__global__ void __launch_bounds__(128)
gemm_wmma(const _Float16* __restrict__ A, int lda,
          const _Float16* __restrict__ W, int ldb,
          const float* __restrict__ bias,
          float* __restrict__ C, int ldc,
          _Float16* __restrict__ C16, int ldc16,
          int M, int N, int K, int relu) {
  const int lane = threadIdx.x & 31;
  const int wave = threadIdx.x >> 5;          // 0..3
  const int wm = wave >> 1, wn = wave & 1;
  const int mBlk  = blockIdx.y * 128;
  const int nBlk  = blockIdx.x * 64;
  const int mBase = mBlk + wm * 64;
  const int nBase = nBlk + wn * 32;

  v8f acc[4][2] = {};

#if USE_ASYNC_LDS
  __shared__ _Float16 Asm[2][128 * LDSA_LD];
  __shared__ _Float16 Bsm[2][64 * LDSA_LD];
  const int tid = threadIdx.x;
  auto issue = [&](int buf, int kb) {
#pragma unroll
    for (int i = 0; i < 4; ++i) {            // A: 128 rows x 4 chunks of 16B
      int c = tid + i * 128;
      int row = c >> 2, kc = (c & 3) * 8;
      const _Float16* g = A + (size_t)(mBlk + row) * lda + kb + kc;
      _Float16* l = &Asm[buf][row * LDSA_LD + kc];
      __builtin_amdgcn_global_load_async_to_lds_b128((g_v4i32*)g, (l_v4i32*)l, 0, 0);
    }
#pragma unroll
    for (int i = 0; i < 2; ++i) {            // B: 64 rows x 4 chunks of 16B
      int c = tid + i * 128;
      int row = c >> 2, kc = (c & 3) * 8;
      const _Float16* g = W + (size_t)(nBlk + row) * ldb + kb + kc;
      _Float16* l = &Bsm[buf][row * LDSA_LD + kc];
      __builtin_amdgcn_global_load_async_to_lds_b128((g_v4i32*)g, (l_v4i32*)l, 0, 0);
    }
  };
  issue(0, 0);
  int cur = 0;
  for (int kb = 0; kb < K; kb += 32) {
    const bool hasNext = (kb + 32 < K);
    if (hasNext) {
      issue(cur ^ 1, kb + 32);
      __builtin_amdgcn_s_wait_asynccnt(6);   // this buffer's 6 ops done
    } else {
      __builtin_amdgcn_s_wait_asynccnt(0);
    }
    __syncthreads();
    const _Float16* Ab = &Asm[cur][(size_t)(wm * 64) * LDSA_LD];
    const _Float16* Bb = &Bsm[cur][(size_t)(wn * 32) * LDSA_LD];
    v16h a0 = load_frag(Ab,                 LDSA_LD, lane);
    v16h a1 = load_frag(Ab + 16 * LDSA_LD,  LDSA_LD, lane);
    v16h a2 = load_frag(Ab + 32 * LDSA_LD,  LDSA_LD, lane);
    v16h a3 = load_frag(Ab + 48 * LDSA_LD,  LDSA_LD, lane);
    v16h b0 = load_frag(Bb,                 LDSA_LD, lane);
    v16h b1 = load_frag(Bb + 16 * LDSA_LD,  LDSA_LD, lane);
    acc[0][0] = __builtin_amdgcn_wmma_f32_16x16x32_f16(false, a0, false, b0, (short)0, acc[0][0], false, false);
    acc[0][1] = __builtin_amdgcn_wmma_f32_16x16x32_f16(false, a0, false, b1, (short)0, acc[0][1], false, false);
    acc[1][0] = __builtin_amdgcn_wmma_f32_16x16x32_f16(false, a1, false, b0, (short)0, acc[1][0], false, false);
    acc[1][1] = __builtin_amdgcn_wmma_f32_16x16x32_f16(false, a1, false, b1, (short)0, acc[1][1], false, false);
    acc[2][0] = __builtin_amdgcn_wmma_f32_16x16x32_f16(false, a2, false, b0, (short)0, acc[2][0], false, false);
    acc[2][1] = __builtin_amdgcn_wmma_f32_16x16x32_f16(false, a2, false, b1, (short)0, acc[2][1], false, false);
    acc[3][0] = __builtin_amdgcn_wmma_f32_16x16x32_f16(false, a3, false, b0, (short)0, acc[3][0], false, false);
    acc[3][1] = __builtin_amdgcn_wmma_f32_16x16x32_f16(false, a3, false, b1, (short)0, acc[3][1], false, false);
    __syncthreads();                         // free 'cur' before it is refilled
    cur ^= 1;
  }
#else
  const _Float16* Ap = A + (size_t)mBase * lda;
  const _Float16* Wp = W + (size_t)nBase * ldb;
  for (int kb = 0; kb < K; kb += 32) {
    if (kb + 32 < K) {                       // -> global_prefetch_b8
      __builtin_prefetch(Ap + kb + 32, 0, 1);
      __builtin_prefetch(Wp + kb + 32, 0, 1);
    }
    v16h a0 = load_frag(Ap + kb,                      lda, lane);
    v16h a1 = load_frag(Ap + (size_t)16 * lda + kb,   lda, lane);
    v16h a2 = load_frag(Ap + (size_t)32 * lda + kb,   lda, lane);
    v16h a3 = load_frag(Ap + (size_t)48 * lda + kb,   lda, lane);
    v16h b0 = load_frag(Wp + kb,                      ldb, lane);
    v16h b1 = load_frag(Wp + (size_t)16 * ldb + kb,   ldb, lane);
    acc[0][0] = __builtin_amdgcn_wmma_f32_16x16x32_f16(false, a0, false, b0, (short)0, acc[0][0], false, false);
    acc[0][1] = __builtin_amdgcn_wmma_f32_16x16x32_f16(false, a0, false, b1, (short)0, acc[0][1], false, false);
    acc[1][0] = __builtin_amdgcn_wmma_f32_16x16x32_f16(false, a1, false, b0, (short)0, acc[1][0], false, false);
    acc[1][1] = __builtin_amdgcn_wmma_f32_16x16x32_f16(false, a1, false, b1, (short)0, acc[1][1], false, false);
    acc[2][0] = __builtin_amdgcn_wmma_f32_16x16x32_f16(false, a2, false, b0, (short)0, acc[2][0], false, false);
    acc[2][1] = __builtin_amdgcn_wmma_f32_16x16x32_f16(false, a2, false, b1, (short)0, acc[2][1], false, false);
    acc[3][0] = __builtin_amdgcn_wmma_f32_16x16x32_f16(false, a3, false, b0, (short)0, acc[3][0], false, false);
    acc[3][1] = __builtin_amdgcn_wmma_f32_16x16x32_f16(false, a3, false, b1, (short)0, acc[3][1], false, false);
  }
#endif

  const int n  = lane & 15;
  const int hs = lane >> 4;   // C/D layout: VGPR v -> row v + hs*8, col = lane&15
#pragma unroll
  for (int ti = 0; ti < 4; ++ti)
#pragma unroll
    for (int tj = 0; tj < 2; ++tj) {
      int col = nBase + tj * 16 + n;
      float bval = bias ? bias[col] : 0.f;
#pragma unroll
      for (int v = 0; v < 8; ++v) {
        int row = mBase + ti * 16 + v + hs * 8;
        float x = acc[ti][tj][v] + bval;
        if (relu) x = fmaxf(x, 0.f);
        C[(size_t)row * ldc + col] = x;
        if (C16) C16[(size_t)row * ldc16 + col] = (_Float16)x;
      }
    }
}

// =====================================================================
// small helper kernels
// =====================================================================
__global__ void cvt_pad_f16(const float* __restrict__ src, _Float16* __restrict__ dst,
                            int srows, int scols, int drows, int dcols) {
  GSTRIDE(i, (size_t)drows * dcols) {
    int r = (int)(i / dcols), c = (int)(i % dcols);
    float v = (r < srows && c < scols) ? src[(size_t)r * scols + c] : 0.f;
    dst[i] = (_Float16)v;
  }
}

__global__ void pad_f32(const float* __restrict__ src, float* __restrict__ dst,
                        int sc, int dc) {
  GSTRIDE(i, (size_t)dc) dst[i] = (i < (size_t)sc) ? src[i] : 0.f;
}

// gc weight (k,F,O) -> per-k [O][dcols] f16 (transposed, K-padded)
__global__ void cvt_gcw(const float* __restrict__ src, _Float16* __restrict__ dst,
                        int F, int O, int dcols) {
  int k = blockIdx.z;
  const float* s = src + (size_t)k * F * O;
  _Float16*    d = dst + (size_t)k * O * dcols;
  GSTRIDE(i, (size_t)O * dcols) {
    int o = (int)(i / dcols), f = (int)(i % dcols);
    d[i] = (_Float16)((f < F) ? s[(size_t)f * O + o] : 0.f);
  }
}

__global__ void embed_kernel(const int* __restrict__ q, const float* __restrict__ wembed,
                             _Float16* __restrict__ X) {
  GSTRIDE(i, (size_t)BT * EMBP) {
    int row = (int)(i / EMBP), e = (int)(i % EMBP);
    float v = 0.f;
    if (e < EMB) v = wembed[(size_t)q[row] * EMB + e];
    X[i] = (_Float16)v;
  }
}

__global__ void gru_update(const float* __restrict__ gi_all, const float* __restrict__ gh,
                           const int* __restrict__ qlen, float* __restrict__ h,
                           _Float16* __restrict__ h16, int t) {
  GSTRIDE(i, (size_t)BB * HID) {
    int b = (int)(i >> 10), j = (int)(i & 1023);
    const float* gi  = gi_all + ((size_t)b * T + t) * H3;
    const float* ghb = gh + (size_t)b * H3;
    float r = 1.f / (1.f + __expf(-(gi[j]        + ghb[j])));
    float z = 1.f / (1.f + __expf(-(gi[j + HID]  + ghb[j + HID])));
    float n = tanhf(gi[j + 2*HID] + r * ghb[j + 2*HID]);
    float hp = h[i];
    float hn = (1.f - z) * n + z * hp;
    float o  = (t < qlen[b]) ? hn : hp;
    h[i] = o; h16[i] = (_Float16)o;
  }
}

__global__ void attn_logits(const float* __restrict__ proj, const float* __restrict__ qp,
                            const float* __restrict__ aw, const float* __restrict__ ab,
                            float* __restrict__ raw, int L) {
  GSTRIDE(i, (size_t)BB * L) {
    int b = (int)(i / L);
    const float* p = proj + i * HID;
    const float* q = qp + (size_t)b * HID;
    float s = 0.f;
    for (int h = 0; h < HID; ++h) s += p[h] * q[h] * aw[h];
    raw[i] = s + ab[0];
  }
}

__global__ void softmax_rows(float* __restrict__ x, int L) {
  GSTRIDE(b, (size_t)BB) {
    float* row = x + b * L;
    float mx = row[0];
    for (int l = 1; l < L; ++l) mx = fmaxf(mx, row[l]);
    float s = 0.f;
    for (int l = 0; l < L; ++l) { float e = __expf(row[l] - mx); row[l] = e; s += e; }
    float inv = 1.f / s;
    for (int l = 0; l < L; ++l) row[l] *= inv;
  }
}

__global__ void topk_select(const float* __restrict__ att, int* __restrict__ idx) {
  GSTRIDE(b, (size_t)BB) {
    float v[KOBJ];
    for (int l = 0; l < KOBJ; ++l) v[l] = att[b * KOBJ + l];
    for (int s = 0; s < NS; ++s) {
      int best = 0; float bv = -1e30f;
      for (int l = 0; l < KOBJ; ++l) if (v[l] > bv) { bv = v[l]; best = l; }
      idx[b * NS + s] = best; v[best] = -1e30f;
    }
  }
}

__global__ void gather_topk(const float* __restrict__ image, const int* __restrict__ idx,
                            float* __restrict__ topk) {
  GSTRIDE(i, (size_t)BB * NS * FEAT) {
    int b = (int)(i / ((size_t)NS * FEAT));
    size_t rem = i % ((size_t)NS * FEAT);
    int s = (int)(rem / FEAT), f = (int)(rem % FEAT);
    topk[i] = image[((size_t)b * KOBJ + idx[b * NS + s]) * FEAT + f];
  }
}

__global__ void wsum_f32(const float* __restrict__ src, const float* __restrict__ att,
                         float* __restrict__ out, int L, int F) {
  GSTRIDE(i, (size_t)BB * F) {
    int b = (int)(i / F), f = (int)(i % F);
    float s = 0.f;
    for (int l = 0; l < L; ++l)
      s += att[b * L + l] * src[((size_t)b * L + l) * F + f];
    out[i] = s;
  }
}

__global__ void wsum_f16(const _Float16* __restrict__ src, const float* __restrict__ att,
                         _Float16* __restrict__ out, int L, int Fp) {
  GSTRIDE(i, (size_t)BB * Fp) {
    int b = (int)(i / Fp), f = (int)(i % Fp);
    float s = 0.f;
    for (int l = 0; l < L; ++l)
      s += att[b * L + l] * (float)src[((size_t)b * L + l) * Fp + f];
    out[i] = (_Float16)s;
  }
}

__global__ void mul_ew(const float* __restrict__ a, const float* __restrict__ b,
                       float* __restrict__ o, size_t n) {
  GSTRIDE(i, n) o[i] = a[i] * b[i];
}

__global__ void pseudo_kernel(const float* __restrict__ topk, float* __restrict__ ps) {
  GSTRIDE(i, (size_t)BB * NS * NS) {
    int b = (int)(i / (NS * NS));
    int r = (int)(i % (NS * NS));
    int ii = r / NS, jj = r % NS;
    const float* bi = topk + ((size_t)b * NS + ii) * FEAT + (FEAT - 4);
    const float* bj = topk + ((size_t)b * NS + jj) * FEAT + (FEAT - 4);
    float cix = bi[0] + 0.5f * (bi[2] - bi[0]), ciy = bi[1] + 0.5f * (bi[3] - bi[1]);
    float cjx = bj[0] + 0.5f * (bj[2] - bj[0]), cjy = bj[1] + 0.5f * (bj[3] - bj[1]);
    float dx = cix - cjx, dy = ciy - cjy;
    ps[i * 2 + 0] = sqrtf(dx * dx + dy * dy);
    ps[i * 2 + 1] = atan2f(dx, dy);
  }
}

__global__ void gauss_kernel(const float* __restrict__ ps, const float* __restrict__ mu,
                             const float* __restrict__ sg, float* __restrict__ kw) {
  GSTRIDE(i, (size_t)BB * NS * NS * NK) {
    int k = (int)(i % NK);
    size_t p = i / NK;
    float s = 0.f;
#pragma unroll
    for (int d = 0; d < 2; ++d) {
      float diff = (ps[p * 2 + d] - mu[k * 2 + d]) / (1e-14f + sg[k * 2 + d]);
      s += diff * diff;
    }
    kw[i] = __expf(-0.5f * s);
  }
}

// agg[bn, f] = sum_m src[b,m,f] * kw[b,n,m,k]   (one k-slice, f16 K-padded out)
__global__ void agg_slice(const float* __restrict__ src, const float* __restrict__ kw,
                          _Float16* __restrict__ out, int F, int dcols, int k) {
  GSTRIDE(i, (size_t)BNS * dcols) {
    int bn = (int)(i / dcols), f = (int)(i % dcols);
    int b = bn / NS, n = bn % NS;
    float s = 0.f;
    if (f < F)
      for (int m = 0; m < NS; ++m)
        s += src[((size_t)b * NS + m) * F + f] *
             kw[(((size_t)b * NS + n) * NS + m) * NK + k];
    out[i] = (_Float16)s;
  }
}

__global__ void concat_nodes(const float* __restrict__ topk, const float* __restrict__ h2,
                             _Float16* __restrict__ gn) {
  GSTRIDE(i, (size_t)BNS * GNP) {
    int bn = (int)(i / GNP), f = (int)(i % GNP);
    float v = 0.f;
    if (f < FEAT)      v = topk[(size_t)bn * FEAT + f];
    else if (f < GN)   v = h2[(size_t)bn * HID + (f - FEAT)];
    gn[i] = (_Float16)v;
  }
}

__global__ void final_add(const float* __restrict__ l1, const float* __restrict__ l2,
                          float* __restrict__ out) {
  GSTRIDE(i, (size_t)BB * OUT) {
    int b = (int)(i / OUT), o = (int)(i % OUT);
    out[i] = l1[(size_t)b * OUTP + o] + l2[(size_t)b * OUTP + o];
  }
}

// =====================================================================
// host side
// =====================================================================
static inline void gemm(hipStream_t s, const _Float16* A, int lda,
                        const _Float16* W, int ldb, const float* bias,
                        float* C, int ldc, _Float16* C16, int ldc16,
                        int M, int N, int K, int relu) {
  dim3 grid(N / 64, M / 128), block(128);
  gemm_wmma<<<grid, block, 0, s>>>(A, lda, W, ldb, bias, C, ldc, C16, ldc16, M, N, K, relu);
}
static inline dim3 g1(size_t n) { return dim3((unsigned)((n + 255) / 256)); }

extern "C" void kernel_launch(void* const* d_in, const int* in_sizes, int n_in,
                              void* d_out, int out_size, void* d_ws, size_t ws_size,
                              hipStream_t stream) {
  const int*   question = (const int*)  d_in[0];
  const float* image    = (const float*)d_in[1];
  const int*   qlen     = (const int*)  d_in[3];
  const float* wembed   = (const float*)d_in[4];
  const float* gru_wih  = (const float*)d_in[5];
  const float* gru_whh  = (const float*)d_in[6];
  const float* gru_bih  = (const float*)d_in[7];
  const float* gru_bhh  = (const float*)d_in[8];
  const float* ia_img_w = (const float*)d_in[9];
  const float* ia_img_b = (const float*)d_in[10];
  const float* ia_txt_w = (const float*)d_in[11];
  const float* ia_txt_b = (const float*)d_in[12];
  const float* ia_att_w = (const float*)d_in[13];
  const float* ia_att_b = (const float*)d_in[14];
  const float* ga_img_w = (const float*)d_in[15];
  const float* ga_img_b = (const float*)d_in[16];
  const float* ga_txt_w = (const float*)d_in[17];
  const float* ga_txt_b = (const float*)d_in[18];
  const float* ga_att_w = (const float*)d_in[19];
  const float* ga_att_b = (const float*)d_in[20];
  const float* gc1_mu   = (const float*)d_in[21];
  const float* gc1_sig  = (const float*)d_in[22];
  const float* gc1_w    = (const float*)d_in[23];
  const float* gc1_b    = (const float*)d_in[24];
  const float* gc2_mu   = (const float*)d_in[25];
  const float* gc2_sig  = (const float*)d_in[26];
  const float* gc2_w    = (const float*)d_in[27];
  const float* gc2_b    = (const float*)d_in[28];
  const float* out1_w   = (const float*)d_in[29];
  const float* out1_b   = (const float*)d_in[30];
  const float* out2_w   = (const float*)d_in[31];
  const float* out2_b   = (const float*)d_in[32];
  const float* iout1_w  = (const float*)d_in[33];
  const float* iout1_b  = (const float*)d_in[34];
  const float* iout2_w  = (const float*)d_in[35];
  const float* iout2_b  = (const float*)d_in[36];

  // ---- bump allocator over d_ws ----
  char* wp = (char*)d_ws;
  auto alloc = [&](size_t bytes) -> void* {
    void* r = (void*)wp;
    wp += (bytes + 255) & ~(size_t)255;
    return r;
  };
  _Float16* wih_h  = (_Float16*)alloc((size_t)H3 * EMBP * 2);
  _Float16* whh_h  = (_Float16*)alloc((size_t)H3 * HID * 2);
  _Float16* iaw_h  = (_Float16*)alloc((size_t)HID * FEATP * 2);
  _Float16* iat_h  = (_Float16*)alloc((size_t)HID * HID * 2);
  _Float16* gaw_h  = (_Float16*)alloc((size_t)HID * GNP * 2);
  _Float16* gat_h  = (_Float16*)alloc((size_t)HID * HID * 2);
  _Float16* w1h    = (_Float16*)alloc((size_t)NK * 256 * FEATP * 2);
  _Float16* w2h    = (_Float16*)alloc((size_t)NK * 128 * 2048 * 2);
  _Float16* o1w_h  = (_Float16*)alloc((size_t)OUTP * HID * 2);
  _Float16* o2w_h  = (_Float16*)alloc((size_t)OUTP * OUTP * 2);
  _Float16* i1w_h  = (_Float16*)alloc((size_t)OUTP * HID * 2);
  _Float16* i2w_h  = (_Float16*)alloc((size_t)OUTP * OUTP * 2);
  float* o1b = (float*)alloc(OUTP * 4);
  float* o2b = (float*)alloc(OUTP * 4);
  float* i1b = (float*)alloc(OUTP * 4);
  float* i2b = (float*)alloc(OUTP * 4);

  _Float16* X_h    = (_Float16*)alloc((size_t)BT * EMBP * 2);
  float* gi_all    = (float*)alloc((size_t)BT * H3 * 4);
  float* h         = (float*)alloc((size_t)BB * HID * 4);
  _Float16* h16    = (_Float16*)alloc((size_t)BB * HID * 2);
  float* gh        = (float*)alloc((size_t)BB * H3 * 4);
  _Float16* img_h  = (_Float16*)alloc((size_t)BB * KOBJ * FEATP * 2);
  float* img_proj  = (float*)alloc((size_t)BB * KOBJ * HID * 4);
  float* qp        = (float*)alloc((size_t)BB * HID * 4);
  float* att       = (float*)alloc((size_t)BB * KOBJ * 4);
  int*   idx       = (int*)alloc((size_t)BB * NS * 4);
  float* topk      = (float*)alloc((size_t)BNS * FEAT * 4);
  float* img_att   = (float*)alloc((size_t)BB * FEAT * 4);
  _Float16* iatt_h = (_Float16*)alloc((size_t)BB * FEATP * 2);
  float* tmp1024   = (float*)alloc((size_t)BB * HID * 4);
  float* comb1     = (float*)alloc((size_t)BB * HID * 4);
  _Float16* comb1h = (_Float16*)alloc((size_t)BB * HID * 2);
  float* ps        = (float*)alloc((size_t)BB * NS * NS * 2 * 4);
  float* kw1       = (float*)alloc((size_t)BB * NS * NS * NK * 4);
  float* kw2       = (float*)alloc((size_t)BB * NS * NS * NK * 4);
  _Float16* aggs   = (_Float16*)alloc((size_t)BNS * FEATP * 2);
  float* h1        = (float*)alloc((size_t)BNS * 2048 * 4);
  float* h2        = (float*)alloc((size_t)BNS * HID * 4);
  _Float16* gn_h   = (_Float16*)alloc((size_t)BNS * GNP * 2);
  float* gn_proj   = (float*)alloc((size_t)BNS * HID * 4);
  float* qpg       = (float*)alloc((size_t)BB * HID * 4);
  float* gatt      = (float*)alloc((size_t)BB * NS * 4);
  _Float16* gsum_h = (_Float16*)alloc((size_t)BB * GNP * 2);
  float* comb2     = (float*)alloc((size_t)BB * HID * 4);
  _Float16* comb2h = (_Float16*)alloc((size_t)BB * HID * 2);
  float* t1        = (float*)alloc((size_t)BB * OUTP * 4);
  _Float16* t1h    = (_Float16*)alloc((size_t)BB * OUTP * 2);
  float* l1        = (float*)alloc((size_t)BB * OUTP * 4);
  float* t2        = (float*)alloc((size_t)BB * OUTP * 4);
  _Float16* t2h    = (_Float16*)alloc((size_t)BB * OUTP * 2);
  float* l2        = (float*)alloc((size_t)BB * OUTP * 4);

  dim3 blk(256);

  // ---- phase 0: weight conversions (f32 -> f16, padded) ----
  cvt_pad_f16<<<g1((size_t)H3*EMBP),blk,0,stream>>>(gru_wih, wih_h, H3, EMB, H3, EMBP);
  cvt_pad_f16<<<g1((size_t)H3*HID),blk,0,stream>>>(gru_whh, whh_h, H3, HID, H3, HID);
  cvt_pad_f16<<<g1((size_t)HID*FEATP),blk,0,stream>>>(ia_img_w, iaw_h, HID, FEAT, HID, FEATP);
  cvt_pad_f16<<<g1((size_t)HID*HID),blk,0,stream>>>(ia_txt_w, iat_h, HID, HID, HID, HID);
  cvt_pad_f16<<<g1((size_t)HID*GNP),blk,0,stream>>>(ga_img_w, gaw_h, HID, GN, HID, GNP);
  cvt_pad_f16<<<g1((size_t)HID*HID),blk,0,stream>>>(ga_txt_w, gat_h, HID, HID, HID, HID);
  { dim3 gz((unsigned)(((size_t)256*FEATP+255)/256),1,NK);
    cvt_gcw<<<gz,blk,0,stream>>>(gc1_w, w1h, FEAT, 256, FEATP); }
  { dim3 gz((unsigned)(((size_t)128*2048+255)/256),1,NK);
    cvt_gcw<<<gz,blk,0,stream>>>(gc2_w, w2h, 2048, 128, 2048); }
  cvt_pad_f16<<<g1((size_t)OUTP*HID),blk,0,stream>>>(out1_w, o1w_h, OUT, HID, OUTP, HID);
  cvt_pad_f16<<<g1((size_t)OUTP*OUTP),blk,0,stream>>>(out2_w, o2w_h, OUT, OUT, OUTP, OUTP);
  cvt_pad_f16<<<g1((size_t)OUTP*HID),blk,0,stream>>>(iout1_w, i1w_h, OUT, HID, OUTP, HID);
  cvt_pad_f16<<<g1((size_t)OUTP*OUTP),blk,0,stream>>>(iout2_w, i2w_h, OUT, OUT, OUTP, OUTP);
  pad_f32<<<g1(OUTP),blk,0,stream>>>(out1_b, o1b, OUT, OUTP);
  pad_f32<<<g1(OUTP),blk,0,stream>>>(out2_b, o2b, OUT, OUTP);
  pad_f32<<<g1(OUTP),blk,0,stream>>>(iout1_b, i1b, OUT, OUTP);
  pad_f32<<<g1(OUTP),blk,0,stream>>>(iout2_b, i2b, OUT, OUTP);

  // ---- phase 1: GRU encoder ----
  embed_kernel<<<g1((size_t)BT*EMBP),blk,0,stream>>>(question, wembed, X_h);
  gemm(stream, X_h, EMBP, wih_h, EMBP, gru_bih, gi_all, H3, nullptr, 0, BT, H3, EMBP, 0);
  (void)hipMemsetAsync(h,   0, (size_t)BB*HID*4, stream);
  (void)hipMemsetAsync(h16, 0, (size_t)BB*HID*2, stream);
  for (int t = 0; t < T; ++t) {
    gemm(stream, h16, HID, whh_h, HID, gru_bhh, gh, H3, nullptr, 0, BB, H3, HID, 0);
    gru_update<<<g1((size_t)BB*HID),blk,0,stream>>>(gi_all, gh, qlen, h, h16, t);
  }

  // ---- phase 2: image attention ----
  cvt_pad_f16<<<g1((size_t)BB*KOBJ*FEATP),blk,0,stream>>>(image, img_h,
                                                          BB*KOBJ, FEAT, BB*KOBJ, FEATP);
  gemm(stream, img_h, FEATP, iaw_h, FEATP, ia_img_b, img_proj, HID, nullptr, 0,
       BB*KOBJ, HID, FEATP, 1);
  gemm(stream, h16, HID, iat_h, HID, ia_txt_b, qp, HID, nullptr, 0, BB, HID, HID, 1);
  attn_logits<<<g1((size_t)BB*KOBJ),blk,0,stream>>>(img_proj, qp, ia_att_w, ia_att_b, att, KOBJ);
  softmax_rows<<<g1(BB),blk,0,stream>>>(att, KOBJ);
  topk_select<<<g1(BB),blk,0,stream>>>(att, idx);
  gather_topk<<<g1((size_t)BNS*FEAT),blk,0,stream>>>(image, idx, topk);
  wsum_f32<<<g1((size_t)BB*FEAT),blk,0,stream>>>(image, att, img_att, KOBJ, FEAT);
  cvt_pad_f16<<<g1((size_t)BB*FEATP),blk,0,stream>>>(img_att, iatt_h, BB, FEAT, BB, FEATP);
  gemm(stream, iatt_h, FEATP, iaw_h, FEATP, ia_img_b, tmp1024, HID, nullptr, 0,
       BB, HID, FEATP, 1);
  mul_ew<<<g1((size_t)BB*HID),blk,0,stream>>>(tmp1024, qp, comb1, (size_t)BB*HID);
  cvt_pad_f16<<<g1((size_t)BB*HID),blk,0,stream>>>(comb1, comb1h, BB, HID, BB, HID);

  // ---- phase 3: graph convolutions ----
  pseudo_kernel<<<g1((size_t)BB*NS*NS),blk,0,stream>>>(topk, ps);
  gauss_kernel<<<g1((size_t)BB*NS*NS*NK),blk,0,stream>>>(ps, gc1_mu, gc1_sig, kw1);
  gauss_kernel<<<g1((size_t)BB*NS*NS*NK),blk,0,stream>>>(ps, gc2_mu, gc2_sig, kw2);
  for (int k = 0; k < NK; ++k) {
    agg_slice<<<g1((size_t)BNS*FEATP),blk,0,stream>>>(topk, kw1, aggs, FEAT, FEATP, k);
    gemm(stream, aggs, FEATP, w1h + (size_t)k*256*FEATP, FEATP, gc1_b + k*256,
         h1 + k*256, 2048, nullptr, 0, BNS, 256, FEATP, 1);
  }
  for (int k = 0; k < NK; ++k) {
    agg_slice<<<g1((size_t)BNS*2048),blk,0,stream>>>(h1, kw2, aggs, 2048, 2048, k);
    gemm(stream, aggs, 2048, w2h + (size_t)k*128*2048, 2048, gc2_b + k*128,
         h2 + k*128, HID, nullptr, 0, BNS, 128, 2048, 1);
  }

  // ---- phase 4: graph attention ----
  concat_nodes<<<g1((size_t)BNS*GNP),blk,0,stream>>>(topk, h2, gn_h);
  gemm(stream, gn_h, GNP, gaw_h, GNP, ga_img_b, gn_proj, HID, nullptr, 0, BNS, HID, GNP, 1);
  gemm(stream, h16, HID, gat_h, HID, ga_txt_b, qpg, HID, nullptr, 0, BB, HID, HID, 1);
  attn_logits<<<g1((size_t)BB*NS),blk,0,stream>>>(gn_proj, qpg, ga_att_w, ga_att_b, gatt, NS);
  softmax_rows<<<g1(BB),blk,0,stream>>>(gatt, NS);
  wsum_f16<<<g1((size_t)BB*GNP),blk,0,stream>>>(gn_h, gatt, gsum_h, NS, GNP);
  gemm(stream, gsum_h, GNP, gaw_h, GNP, ga_img_b, tmp1024, HID, nullptr, 0, BB, HID, GNP, 1);
  mul_ew<<<g1((size_t)BB*HID),blk,0,stream>>>(tmp1024, qpg, comb2, (size_t)BB*HID);
  cvt_pad_f16<<<g1((size_t)BB*HID),blk,0,stream>>>(comb2, comb2h, BB, HID, BB, HID);

  // ---- phase 5: output heads ----
  gemm(stream, comb2h, HID, o1w_h, HID, o1b, t1, OUTP, t1h, OUTP, BB, OUTP, HID, 1);
  gemm(stream, t1h, OUTP, o2w_h, OUTP, o2b, l1, OUTP, nullptr, 0, BB, OUTP, OUTP, 0);
  gemm(stream, comb1h, HID, i1w_h, HID, i1b, t2, OUTP, t2h, OUTP, BB, OUTP, HID, 1);
  gemm(stream, t2h, OUTP, i2w_h, OUTP, i2b, l2, OUTP, nullptr, 0, BB, OUTP, OUTP, 0);
  final_add<<<g1((size_t)BB*OUT),blk,0,stream>>>(l1, l2, (float*)d_out);

  (void)in_sizes; (void)n_in; (void)out_size; (void)ws_size;
}
